// AddAttention_154618823089
// MI455X (gfx1250) — compile-verified, matrix operand order checked by reading, not run on hardware
//
#include <hip/hip_runtime.h>

typedef __bf16 bf16;
typedef __attribute__((ext_vector_type(16))) __bf16 v16bf;
typedef __attribute__((ext_vector_type(8)))  __bf16 v8bf;
typedef __attribute__((ext_vector_type(8)))  float  v8f;

#define DIM   1024
#define L_SEQ 4096
#define NROWS 8192      // B*L
#define QPAD  8
#define LDSTR (DIM + QPAD)

__device__ __forceinline__ v8f wmma_bf16(v16bf a, v16bf b, v8f c) {
    return __builtin_amdgcn_wmma_f32_16x16x32_bf16(
        /*neg_a=*/false, a, /*neg_b=*/false, b,
        /*c_mod=*/(short)0, c, /*reuse_a=*/false, /*reuse_b=*/false);
}

// Load a 16x32 bf16 A-style fragment row from fp32 memory.
// lane%16 picks the row (caller passes rowPtr for its row), h = lane/16.
__device__ __forceinline__ v16bf frag_from_f32(const float* __restrict__ rowPtr,
                                               int kk, int h) {
    const float4* p0 = (const float4*)(rowPtr + kk + h * 8);
    const float4* p1 = (const float4*)(rowPtr + kk + 16 + h * 8);
    float4 a0 = p0[0], a1 = p0[1];
    float4 b0 = p1[0], b1 = p1[1];
    v16bf f;
    f[0]=(bf16)a0.x; f[1]=(bf16)a0.y; f[2]=(bf16)a0.z; f[3]=(bf16)a0.w;
    f[4]=(bf16)a1.x; f[5]=(bf16)a1.y; f[6]=(bf16)a1.z; f[7]=(bf16)a1.w;
    f[8]=(bf16)b0.x; f[9]=(bf16)b0.y; f[10]=(bf16)b0.z; f[11]=(bf16)b0.w;
    f[12]=(bf16)b1.x; f[13]=(bf16)b1.y; f[14]=(bf16)b1.z; f[15]=(bf16)b1.w;
    return f;
}

// Same fragment from bf16 memory (global or LDS).
__device__ __forceinline__ v16bf frag_from_bf16(const bf16* __restrict__ rowPtr,
                                                int kk, int h) {
    v8bf lo = *(const v8bf*)(rowPtr + kk + h * 8);
    v8bf hi = *(const v8bf*)(rowPtr + kk + 16 + h * 8);
    return __builtin_shufflevector(lo, hi, 0,1,2,3,4,5,6,7,8,9,10,11,12,13,14,15);
}

// gfx1250 async global->LDS copy (ASYNCcnt-tracked, no VGPR staging).
__device__ __forceinline__ void async_g2l_b128(unsigned lds_off, const void* gaddr) {
    asm volatile("global_load_async_to_lds_b128 %0, %1, off"
                 :: "v"(lds_off), "v"(gaddr) : "memory");
}
__device__ __forceinline__ void wait_asynccnt0() {
    asm volatile("s_wait_asynccnt 0x0" ::: "memory");
}

// ---------------------------------------------------------------------------
// Kernel 1: Q/K projection (bf16 WMMA, fp32 accumulate) + fused RoPE epilogue.
// One wave per (16-row tile, 16-column pair {n, n+512}).
// ---------------------------------------------------------------------------
__global__ void proj_rope_kernel(const float* __restrict__ hidden,
                                 const float* __restrict__ Wq,
                                 const float* __restrict__ Wk,
                                 bf16* __restrict__ Qo,
                                 bf16* __restrict__ Ko) {
    const int lane    = threadIdx.x;
    const int r       = lane & 15;
    const int h       = lane >> 4;
    const int rowBase = blockIdx.x * 16;        // 0..8191 step 16
    const int nLo     = blockIdx.y * 16;        // 0..511 step 16
    const int nHi     = nLo + 512;

    const float* aRow  = hidden + (size_t)(rowBase + r) * DIM;
    const float* bqLo  = Wq + (size_t)(nLo + r) * DIM;
    const float* bqHi  = Wq + (size_t)(nHi + r) * DIM;
    const float* bkLo  = Wk + (size_t)(nLo + r) * DIM;
    const float* bkHi  = Wk + (size_t)(nHi + r) * DIM;

    v8f aqLo = {}; v8f aqHi = {}; v8f akLo = {}; v8f akHi = {};

    for (int kk = 0; kk < DIM; kk += 32) {
        v16bf a  = frag_from_f32(aRow, kk, h);
        v16bf q0 = frag_from_f32(bqLo, kk, h);
        v16bf q1 = frag_from_f32(bqHi, kk, h);
        v16bf k0 = frag_from_f32(bkLo, kk, h);
        v16bf k1 = frag_from_f32(bkHi, kk, h);
        aqLo = wmma_bf16(a, q0, aqLo);
        aqHi = wmma_bf16(a, q1, aqHi);
        akLo = wmma_bf16(a, k0, akLo);
        akHi = wmma_bf16(a, k1, akHi);
    }

    // RoPE: column pair (n, n+512) shares cos/sin; freq index m = n (< 512).
    const int   m    = nLo + r;
    const float invf = exp2f(-5.0f * (float)m / 512.0f);   // 32^(-m/512)

#pragma unroll
    for (int v = 0; v < 8; ++v) {
        const int M   = v + 8 * h;                 // row within tile (C layout)
        const int row = rowBase + M;
        const int pos = row & (L_SEQ - 1);
        float ang = (float)pos * invf;
        float c = (float)(bf16)__cosf(ang);        // reference rounds cos/sin to bf16
        float s = (float)(bf16)__sinf(ang);

        float qlo = (float)(bf16)aqLo[v];          // projection rounded to bf16 first
        float qhi = (float)(bf16)aqHi[v];
        float klo = (float)(bf16)akLo[v];
        float khi = (float)(bf16)akHi[v];

        size_t base = (size_t)row * DIM;
        Qo[base + nLo + r] = (bf16)(qlo * c - qhi * s);
        Qo[base + nHi + r] = (bf16)(qhi * c + qlo * s);
        Ko[base + nLo + r] = (bf16)(klo * c - khi * s);
        Ko[base + nHi + r] = (bf16)(khi * c + klo * s);
    }
}

// ---------------------------------------------------------------------------
// Kernel 2: zero the row-sum accumulator (workspace is poisoned by harness).
// ---------------------------------------------------------------------------
__global__ void zero_kernel(float* __restrict__ p, int n) {
    int i = blockIdx.x * 256 + threadIdx.x;
    if (i < n) p[i] = 0.0f;
}

// ---------------------------------------------------------------------------
// Kernel 3: flash-style causal exp-score row sums.
// o[b, q] = sum_{k<=q} exp(q_vec . k_vec / 32)
// 4 waves per block share one async-staged Q tile in LDS; each wave strides
// over key tiles within a 64-tile chunk.
// ---------------------------------------------------------------------------
__global__ void flash_rowsum_kernel(const bf16* __restrict__ Q,
                                    const bf16* __restrict__ K,
                                    float* __restrict__ o) {
    __shared__ bf16 qlds[16 * LDSTR];

    const int t    = threadIdx.x;         // 0..127
    const int lane = t & 31;
    const int w    = t >> 5;              // wave id 0..3
    const int qt   = blockIdx.x;          // 0..255
    const int b    = blockIdx.y;          // 0..1
    const int kc   = blockIdx.z;          // 0..3  (chunks of 64 key tiles)
    const int ktBegin = kc * 64;
    if (ktBegin > qt) return;             // uniform: entirely above the diagonal
    const int ktEnd = (ktBegin + 64 < qt + 1) ? (ktBegin + 64) : (qt + 1);

    // Async-stage the 16x1024 bf16 Q tile into LDS (padded stride LDSTR).
    // 2048 b128 chunks, 16 per thread, ASYNCcnt-tracked.
    {
        const bf16* qbase = Q + (size_t)(b * L_SEQ + qt * 16) * DIM;
        const unsigned lds0 = (unsigned)(size_t)(void*)qlds;
#pragma unroll
        for (int i = 0; i < 16; ++i) {
            const int c      = t + 128 * i;        // b128 chunk id 0..2047
            const int row    = c >> 7;             // 128 chunks per 2048B row
            const int byteIn = (c & 127) * 16;     // byte offset within row
            const unsigned ldsOff = lds0 + (unsigned)(row * LDSTR * 2 + byteIn);
            const void* g = (const char*)qbase + (size_t)row * DIM * 2 + byteIn;
            async_g2l_b128(ldsOff, g);
        }
        wait_asynccnt0();
        __syncthreads();
    }

    const int r = lane & 15;
    const int h = lane >> 4;
    const bf16* aRow = qlds + r * LDSTR;

    v8f sumExp = {};
    for (int kt = ktBegin + w; kt < ktEnd; kt += 4) {
        const bf16* kRow = K + ((size_t)(b * L_SEQ + kt * 16 + r)) * DIM;
        v8f acc = {};
        for (int kk = 0; kk < DIM; kk += 32) {
            v16bf a  = frag_from_bf16(aRow, kk, h);
            v16bf bb = frag_from_bf16(kRow, kk, h);
            acc = wmma_bf16(a, bb, acc);
        }
#pragma unroll
        for (int v = 0; v < 8; ++v) {
            float e = __expf(acc[v] * (1.0f / 32.0f));
            if (kt == qt) {                       // causal mask on diagonal tile
                const int M = v + 8 * h;          // query row within tile
                if (r > M) e = 0.0f;              // r = key column within tile
            }
            sumExp[v] += e;
        }
    }

    // Reduce across the 16 key columns (each half of the wave independently).
#pragma unroll
    for (int v = 0; v < 8; ++v) {
        float s = sumExp[v];
        s += __shfl_xor(s, 1);
        s += __shfl_xor(s, 2);
        s += __shfl_xor(s, 4);
        s += __shfl_xor(s, 8);
        if (r == 0) {
            const int M = v + 8 * h;
            atomicAdd(&o[b * L_SEQ + qt * 16 + M], s);
        }
    }
}

// ---------------------------------------------------------------------------
// Kernel 4: MLP epilogue. out[i, :] = relu(o[i]*w1 + b1) @ w2^T + b2
// One block per row; each thread produces 4 of the 1024 outputs.
// ---------------------------------------------------------------------------
__global__ void mlp_kernel(const float* __restrict__ o,
                           const float* __restrict__ fc1_w,
                           const float* __restrict__ fc1_b,
                           const float* __restrict__ fc2_w,
                           const float* __restrict__ fc2_b,
                           float* __restrict__ out) {
    const int i = blockIdx.x;             // row 0..8191
    const int t = threadIdx.x;            // 0..255
    const float ov = o[i];

    float hh[16];
#pragma unroll
    for (int j = 0; j < 16; ++j) {
        float x = ov * fc1_w[j] + fc1_b[j];
        hh[j] = x > 0.0f ? x : 0.0f;
    }

#pragma unroll
    for (int q = 0; q < 4; ++q) {
        const int d = t + 256 * q;
        const float4* wv = (const float4*)(fc2_w + (size_t)d * 16);
        float4 w0 = wv[0], w1 = wv[1], w2 = wv[2], w3 = wv[3];
        float acc = fc2_b[d];
        acc += hh[0]*w0.x + hh[1]*w0.y + hh[2]*w0.z + hh[3]*w0.w;
        acc += hh[4]*w1.x + hh[5]*w1.y + hh[6]*w1.z + hh[7]*w1.w;
        acc += hh[8]*w2.x + hh[9]*w2.y + hh[10]*w2.z + hh[11]*w2.w;
        acc += hh[12]*w3.x + hh[13]*w3.y + hh[14]*w3.z + hh[15]*w3.w;
        out[(size_t)i * DIM + d] = acc;
    }
}

extern "C" void kernel_launch(void* const* d_in, const int* in_sizes, int n_in,
                              void* d_out, int out_size, void* d_ws, size_t ws_size,
                              hipStream_t stream) {
    (void)in_sizes; (void)n_in; (void)out_size; (void)ws_size;
    const float* hidden = (const float*)d_in[0];
    const float* Wq     = (const float*)d_in[1];
    const float* Wk     = (const float*)d_in[2];
    const float* fc1_w  = (const float*)d_in[3];
    const float* fc1_b  = (const float*)d_in[4];
    const float* fc2_w  = (const float*)d_in[5];
    const float* fc2_b  = (const float*)d_in[6];
    float* out = (float*)d_out;

    // Workspace layout: Q(16MB) | K(16MB) | o(32KB)
    char* ws = (char*)d_ws;
    bf16*  Qws = (bf16*)ws;
    bf16*  Kws = (bf16*)(ws + (size_t)NROWS * DIM * sizeof(bf16));
    float* ows = (float*)(ws + 2 * (size_t)NROWS * DIM * sizeof(bf16));

    dim3 g1(NROWS / 16, 512 / 16);                  // 512 row tiles x 32 col pairs
    proj_rope_kernel<<<g1, 32, 0, stream>>>(hidden, Wq, Wk, Qws, Kws);

    zero_kernel<<<(NROWS + 255) / 256, 256, 0, stream>>>(ows, NROWS);

    dim3 g2(L_SEQ / 16, 2, 4);                      // 256 q-tiles x 2 batches x 4 k-chunks
    flash_rowsum_kernel<<<g2, 128, 0, stream>>>(Qws, Kws, ows);

    mlp_kernel<<<NROWS, 256, 0, stream>>>(ows, fc1_w, fc1_b, fc2_w, fc2_b, out);
}
